// MiniMaxText01LightningAttention_19430432047128
// MI455X (gfx1250) — matrix operand order repaired and checked by reading, not compiled, not run on hardware
//
#include <hip/hip_runtime.h>
#include <cstdint>
#include <cstddef>

typedef _Float16 half_t;
typedef half_t v16h __attribute__((ext_vector_type(16)));
typedef float  v8f  __attribute__((ext_vector_type(8)));
typedef float  v4f  __attribute__((ext_vector_type(4)));

#define DEVINL __device__ __forceinline__

// ---------- problem constants ----------
constexpr int BATCHC = 2;
constexpr int SEQC   = 4096;
constexpr int HIDC   = 2048;      // K for all three GEMMs
constexpr int HC     = 32;        // heads
constexpr int DC     = 64;        // head dim
constexpr int BBLK   = 256;       // attention block
constexpr int NBLK   = SEQC / BBLK;       // 16
constexpr int MTOT   = BATCHC * SEQC;     // 8192 rows
constexpr int NQKV   = 3 * HC * DC;       // 6144
constexpr int HD     = HC * DC;           // 2048

// ---------- WMMA fragment helpers (gfx1250 wave32 layouts) ----------
DEVINL v16h frag2(const half_t* p0, const half_t* p1) {
  union { v16h v; v4f f[2]; } u;
  u.f[0] = *(const v4f*)p0;
  u.f[1] = *(const v4f*)p1;
  return u.v;
}
DEVINL v16h load_a(const half_t* row, int hs, int k0) {
  return frag2(row + k0 + hs * 8, row + k0 + 16 + hs * 8);
}
DEVINL v16h load_b(const half_t* colT, int hs, int k0) {
  const half_t* p = colT + k0 + hs * 8;
  return frag2(p, p + 16);
}
DEVINL v8f wmma_f16(v16h a, v16h b, v8f c) {
  return __builtin_amdgcn_wmma_f32_16x16x32_f16(false, a, false, b, (short)0, c,
                                                false, false);
}

// ---------- CDNA5 async global->LDS (ASYNCcnt-tracked DMA path) ----------
DEVINL void async_copy16(void* lds, const void* g) {
  asm volatile("global_load_async_to_lds_b128 %0, %1, off"
               :
               : "v"((uint32_t)(uintptr_t)lds), "v"((uint64_t)(uintptr_t)g)
               : "memory");
}
template <int N>
DEVINL void wait_async_n() {
  asm volatile("s_wait_asynccnt %0" ::"n"(N) : "memory");
}

// ---------- f32 -> f16 convert (row-major copy) ----------
__global__ void cvt_kernel(const float* __restrict__ src,
                           half_t* __restrict__ dst, size_t n) {
  size_t i = (size_t)blockIdx.x * blockDim.x + threadIdx.x;
  size_t stride = (size_t)gridDim.x * blockDim.x;
  for (; i < n; i += stride) dst[i] = (half_t)src[i];
}

// ---------- f32 [K][N] -> f16 [N][K] tiled transpose-convert ----------
__global__ __launch_bounds__(256) void cvtT_kernel(const float* __restrict__ src,
                                                   half_t* __restrict__ dst,
                                                   int K, int N) {
  __shared__ float t[32][33];
  int k0 = blockIdx.y * 32, n0 = blockIdx.x * 32;
  int tx = threadIdx.x & 31, ty = threadIdx.x >> 5;
  for (int r = ty; r < 32; r += 8)
    t[r][tx] = src[(size_t)(k0 + r) * N + n0 + tx];
  __syncthreads();
  for (int r = ty; r < 32; r += 8)
    dst[(size_t)(n0 + r) * K + k0 + tx] = (half_t)t[tx][r];
}

// ---------- tiled WMMA GEMM: C[M x N] = A[M x K] * Bt[N x K]^T, K = 2048 ----
// Double-buffered LDS; tiles DMA'd in with global_load_async_to_lds_b128 and
// overlapped with the WMMAs of the previous K-slice.
// MODE 0: silu; scatter Q,K row-major per head + decay-scaled K^T and V^T
// MODE 1: sigmoid -> o0 (f16, M x N)
// MODE 2: identity -> of (f32, M x N)
template <int MODE>
__global__ __launch_bounds__(256) void gemm_kernel(
    const half_t* __restrict__ A, const half_t* __restrict__ Bt, int N,
    half_t* __restrict__ o0, half_t* __restrict__ o1, half_t* __restrict__ o2,
    half_t* __restrict__ o3, float* __restrict__ of) {
  constexpr int LDSROW = 40;  // 32 K-halfs + 8 pad (80B rows, 16B aligned runs)
  constexpr int NK = HIDC / 32;  // 64 K-slices
  __shared__ half_t sA[2][128 * LDSROW];
  __shared__ half_t sBt[2][128 * LDSROW];

  const int tid = threadIdx.x;
  const int lane = tid & 31, w = tid >> 5;
  const int hs = lane >> 4, ln = lane & 15;
  const int m0 = blockIdx.y * 128, n0 = blockIdx.x * 128;
  const int wr = w >> 2, wc = w & 3;  // 2 x 4 wave grid; wave tile 64 x 32

  v8f acc[4][2] = {};

  const int srow = tid >> 1, sko = (tid & 1) * 16;  // staging assignment
  const half_t* gA = A + (size_t)(m0 + srow) * HIDC + sko;
  const half_t* gB = Bt + (size_t)(n0 + srow) * HIDC + sko;
  auto stage = [&](int buf, int k0) {
    async_copy16(sA[buf] + srow * LDSROW + sko, gA + k0);
    async_copy16(sA[buf] + srow * LDSROW + sko + 8, gA + k0 + 8);
    async_copy16(sBt[buf] + srow * LDSROW + sko, gB + k0);
    async_copy16(sBt[buf] + srow * LDSROW + sko + 8, gB + k0 + 8);
  };

  stage(0, 0);
  for (int ki = 0; ki < NK; ++ki) {
    const int cur = ki & 1;
    if (ki + 1 < NK) {
      stage(cur ^ 1, (ki + 1) * 32);  // DMA next slice while computing this one
      wait_async_n<4>();
    } else {
      wait_async_n<0>();
    }
    __syncthreads();

    v16h af[4], bf[2];
#pragma unroll
    for (int tr = 0; tr < 4; ++tr)
      af[tr] = load_a(sA[cur] + (wr * 64 + tr * 16 + ln) * LDSROW, hs, 0);
#pragma unroll
    for (int tc = 0; tc < 2; ++tc)
      bf[tc] = load_b(sBt[cur] + (wc * 32 + tc * 16 + ln) * LDSROW, hs, 0);
#pragma unroll
    for (int tr = 0; tr < 4; ++tr)
#pragma unroll
      for (int tc = 0; tc < 2; ++tc)
        acc[tr][tc] = wmma_f16(af[tr], bf[tc], acc[tr][tc]);
    __syncthreads();  // all waves done with buf[cur] before it is re-DMA'd
  }

  // epilogue: C element i -> M = i + hs*8 (in-tile), N = ln
#pragma unroll
  for (int tr = 0; tr < 4; ++tr)
#pragma unroll
    for (int tc = 0; tc < 2; ++tc) {
      int mbase = m0 + wr * 64 + tr * 16 + hs * 8;  // 8 consecutive rows
      int nbase = n0 + wc * 32 + tc * 16;           // tile-uniform
      if (MODE == 0) {
        int hh = nbase / 192, rem = nbase % 192;
        int which = rem >> 6;  // uniform per tile (16-wide tile in 64 block)
        int d = (rem & 63) + ln;
        int bb = mbase >> 12, ss0 = mbase & 4095;
        size_t rowbase = ((size_t)bb * HC + hh);
        if (which == 0) {  // Q row-major [bh][s][d]
#pragma unroll
          for (int i = 0; i < 8; ++i) {
            float x = acc[tr][tc][i];
            o0[(rowbase * SEQC + ss0 + i) * DC + d] =
                (half_t)(x / (1.0f + __expf(-x)));
          }
        } else if (which == 1) {  // K row-major + decay-scaled K^T
          float sd = exp2f(-0.25f * (float)(hh + 1)) * 0.72728276f;
          float kd = __expf(-sd * (float)(255 - (ss0 & 255)));
          float rr = __expf(sd);
          union { half_t h[8]; v4f f; } u;
#pragma unroll
          for (int i = 0; i < 8; ++i) {
            float x = acc[tr][tc][i];
            float sv = x / (1.0f + __expf(-x));
            o1[(rowbase * SEQC + ss0 + i) * DC + d] = (half_t)sv;
            u.h[i] = (half_t)(sv * kd);
            kd *= rr;
          }
          *(v4f*)(o2 + (rowbase * DC + d) * SEQC + ss0) = u.f;
        } else {  // V^T [bh][d][s]
          union { half_t h[8]; v4f f; } u;
#pragma unroll
          for (int i = 0; i < 8; ++i) {
            float x = acc[tr][tc][i];
            u.h[i] = (half_t)(x / (1.0f + __expf(-x)));
          }
          *(v4f*)(o3 + (rowbase * DC + d) * SEQC + ss0) = u.f;
        }
      } else if (MODE == 1) {
#pragma unroll
        for (int i = 0; i < 8; ++i) {
          float x = acc[tr][tc][i];
          o0[(size_t)(mbase + i) * N + nbase + ln] =
              (half_t)(1.0f / (1.0f + __expf(-x)));
        }
      } else {
#pragma unroll
        for (int i = 0; i < 8; ++i)
          of[(size_t)(mbase + i) * N + nbase + ln] = acc[tr][tc][i];
      }
    }
}

// ---------- lightning attention scan: one workgroup (16 waves) per (b, h) ----
constexpr int SSTR = 264;   // 256 + 8 pad halfs (16B-aligned 8-half runs)
constexpr int KVSTR = 72;   // 64 + 8 pad
constexpr int ATHREADS = 512;  // 16 waves -> 4 per SIMD32

__global__ __launch_bounds__(ATHREADS) void attn_kernel(
    const half_t* __restrict__ Q, const half_t* __restrict__ K,
    const half_t* __restrict__ KTs, const half_t* __restrict__ VT,
    float* __restrict__ attn) {
  extern __shared__ char smem[];
  half_t* sS   = (half_t*)smem;          // 256 x SSTR : masked scores (f16)
  half_t* sVt  = sS + 256 * SSTR;        // 64 x SSTR  : V^T  [e][j]
  half_t* sKt  = sVt + 64 * SSTR;        // 64 x SSTR  : (K*kd)^T [d][j]
  half_t* sKVt = sKt + 64 * SSTR;        // 64 x KVSTR : KV^T f16 [e][d]
  float*  sKV  = (float*)(sKVt + 64 * KVSTR);  // 64 x 64 f32 state [d][e]

  const int tid = threadIdx.x;
  const int lane = tid & 31, w = tid >> 5;  // 16 waves
  const int hs = lane >> 4, ln = lane & 15;
  const int bh = blockIdx.x;
  const int b = bh >> 5, h = bh & 31;

  // decay rate: s = (2^(-8/H))^(h+1) * (1 - L/(NL-1) + 1e-5)
  const float s = exp2f(-0.25f * (float)(h + 1)) * 0.72728276f;
  const float bd = __expf(-s * 256.0f);
  // factored in-tile decay: exp(-s*(gi-gj)) = exp(-16s*(ti-tj)) * eqi * ekj[ii]
  const float eqi = __expf(-s * (float)ln);
  float ekj[8];
#pragma unroll
  for (int ii = 0; ii < 8; ++ii) ekj[ii] = __expf(s * (float)(hs * 8 + ii));

  const half_t* qh = Q + (size_t)bh * SEQC * DC;
  const half_t* kh = K + (size_t)bh * SEQC * DC;

  for (int i = tid; i < 64 * 64; i += ATHREADS) sKV[i] = 0.0f;
  // the one above-diagonal tile per even row-band is constant zero: write once
  if (w < 8) {
    int ti = w * 2;
    union { half_t h[8]; v4f f; } z;
#pragma unroll
    for (int ii = 0; ii < 8; ++ii) z.h[ii] = (half_t)0.0f;
    *(v4f*)(sS + (ti * 16 + ln) * SSTR + (ti + 1) * 16 + hs * 8) = z.f;
  }
  __syncthreads();

  // staging: thread = (row d, 1/8 of the 256 j's) ; 4 chunks per array
  const int crow = tid >> 3, cpart = tid & 7;
  const size_t gtrow = ((size_t)bh * DC + crow) * SEQC + cpart * 32;
  half_t* ldsK = sKt + crow * SSTR + cpart * 32;
  half_t* ldsV = sVt + crow * SSTR + cpart * 32;

  for (int nb = 0; nb < NBLK; ++nb) {
    const half_t* qblk = qh + (size_t)nb * BBLK * DC;
    const half_t* kblk = kh + (size_t)nb * BBLK * DC;

    // async-DMA the pre-transposed, pre-decayed K^T / V^T tiles into LDS
    {
      const half_t* gK = KTs + gtrow + nb * BBLK;
      const half_t* gV = VT + gtrow + nb * BBLK;
#pragma unroll
      for (int c = 0; c < 4; ++c) {
        async_copy16(ldsK + c * 8, gK + c * 8);
        async_copy16(ldsV + c * 8, gV + c * 8);
      }
    }
    // overlap: convert KV state -> transposed f16 for the o_inter WMMA
    for (int i = tid; i < 64 * 64; i += ATHREADS) {
      int d = i >> 6, e = i & 63;
      sKVt[e * KVSTR + d] = (half_t)sKV[i];
    }
    wait_async_n<0>();
    __syncthreads();

    // ---- phase 1: 136 causal score^T tiles round-robined over 16 waves
#pragma unroll 1
    for (int idx = w; idx < 136; idx += 16) {
      int ti = (int)((sqrtf((float)(8 * idx + 1)) - 1.0f) * 0.5f);
      while ((ti + 1) * (ti + 2) / 2 <= idx) ++ti;
      while (ti * (ti + 1) / 2 > idx) --ti;
      int tj = idx - ti * (ti + 1) / 2;
      const half_t* qrow = qblk + (ti * 16 + ln) * DC;
      v16h bq0 = load_b(qrow, hs, 0);
      v16h bq1 = load_b(qrow, hs, 32);
      const half_t* krow = kblk + (tj * 16 + ln) * DC;
      v8f acc = {};
      acc = wmma_f16(load_a(krow, hs, 0), bq0, acc);
      acc = wmma_f16(load_a(krow, hs, 32), bq1, acc);
      float fr = __expf(-16.0f * s * (float)(ti - tj)) * eqi;
      union { half_t h[8]; v4f f; } u;
      if (ti == tj) {  // diagonal tile: causal compare
#pragma unroll
        for (int ii = 0; ii < 8; ++ii)
          u.h[ii] =
              (half_t)((ln >= hs * 8 + ii) ? acc[ii] * fr * ekj[ii] : 0.0f);
      } else {
#pragma unroll
        for (int ii = 0; ii < 8; ++ii)
          u.h[ii] = (half_t)(acc[ii] * fr * ekj[ii]);
      }
      *(v4f*)(sS + (ti * 16 + ln) * SSTR + tj * 16 + hs * 8) = u.f;
    }
    __syncthreads();  // score bands are consumed by other waves

    // ---- phase 2: 64 output tiles round-robined: o = S@V + (q*qd)@KV
#pragma unroll 1
    for (int idx = w; idx < 64; idx += 16) {
      int ti = idx >> 2, tc = idx & 3;
      int rowib = ti * 16 + ln;
      float qd = __expf(-s * (float)(rowib + 1));
      const half_t* qrow = qblk + rowib * DC;
      v16h aq0 = load_a(qrow, hs, 0);
      v16h aq1 = load_a(qrow, hs, 32);
#pragma unroll
      for (int e = 0; e < 16; ++e) {
        aq0[e] = (half_t)((float)aq0[e] * qd);
        aq1[e] = (half_t)((float)aq1[e] * qd);
      }
      const half_t* kvcol = sKVt + (tc * 16 + ln) * KVSTR;
      v8f acc = {};
      acc = wmma_f16(aq0, load_b(kvcol, hs, 0), acc);
      acc = wmma_f16(aq1, load_b(kvcol, hs, 32), acc);
      const half_t* srowp = sS + rowib * SSTR;
      const half_t* vcol = sVt + (tc * 16 + ln) * SSTR;
      const int nks = (ti >> 1) + 1;  // zero above the causal diagonal
#pragma unroll 1
      for (int ks = 0; ks < nks; ++ks)
        acc = wmma_f16(load_a(srowp, hs, ks * 32), load_b(vcol, hs, ks * 32),
                       acc);
#pragma unroll
      for (int i = 0; i < 8; ++i) {
        int gi = ti * 16 + hs * 8 + i;
        size_t m = (size_t)b * SEQC + (size_t)nb * BBLK + gi;
        attn[m * HD + h * DC + tc * 16 + ln] = acc[i];
      }
    }

    // ---- phase 3: KV = bd*KV + (K*kd)^T @ V ; one 16x16 tile per wave
    {
      int td = w >> 2, te = w & 3;
      v8f acc;
#pragma unroll
      for (int i = 0; i < 8; ++i)
        acc[i] = sKV[(td * 16 + hs * 8 + i) * 64 + te * 16 + ln] * bd;
      const half_t* karow = sKt + (td * 16 + ln) * SSTR;
      const half_t* vcol = sVt + (te * 16 + ln) * SSTR;
#pragma unroll 2
      for (int ks = 0; ks < 8; ++ks)
        acc = wmma_f16(load_a(karow, hs, ks * 32), load_b(vcol, hs, ks * 32),
                       acc);
#pragma unroll
      for (int i = 0; i < 8; ++i)
        sKV[(td * 16 + hs * 8 + i) * 64 + te * 16 + ln] = acc[i];
    }
    __syncthreads();  // KV / staging buffers safe for next block
  }
}

// ---------- RMS norm * norm_weight * gate -> f16 ----------
__global__ __launch_bounds__(256) void norm_gate_kernel(
    const float* __restrict__ attn, const half_t* __restrict__ gate,
    const float* __restrict__ nw, half_t* __restrict__ Y) {
  __shared__ float red[8];
  int m = blockIdx.x, tid = threadIdx.x;
  const float* arow = attn + (size_t)m * HD;
  float ss = 0.0f;
  for (int n = tid; n < HD; n += 256) {
    float v = arow[n];
    ss += v * v;
  }
#pragma unroll
  for (int off = 16; off > 0; off >>= 1) ss += __shfl_down(ss, off, 32);
  if ((tid & 31) == 0) red[tid >> 5] = ss;
  __syncthreads();
  if (tid == 0) {
    float v = 0.0f;
#pragma unroll
    for (int i = 0; i < 8; ++i) v += red[i];
    red[0] = v;
  }
  __syncthreads();
  float scale = rsqrtf(red[0] / (float)HD + 1e-5f);
  for (int n = tid; n < HD; n += 256) {
    float g = (float)gate[(size_t)m * HD + n];
    Y[(size_t)m * HD + n] = (half_t)(arow[n] * scale * nw[n] * g);
  }
}

// ---------- host launcher ----------
extern "C" void kernel_launch(void* const* d_in, const int* in_sizes, int n_in,
                              void* d_out, int out_size, void* d_ws,
                              size_t ws_size, hipStream_t stream) {
  const float* hidden = (const float*)d_in[0];
  const float* w_qkv = (const float*)d_in[1];
  const float* norm_w = (const float*)d_in[2];
  const float* w_gate = (const float*)d_in[3];
  const float* w_out = (const float*)d_in[4];
  float* out = (float*)d_out;

  char* ws = (char*)d_ws;
  auto take = [&](size_t bytes) {
    char* p = ws;
    ws += (bytes + 255) & ~(size_t)255;
    return p;
  };
  half_t* Xh    = (half_t*)take((size_t)MTOT * HIDC * 2);
  half_t* WqkvT = (half_t*)take((size_t)HIDC * NQKV * 2);  // [N][K]
  half_t* WgT   = (half_t*)take((size_t)HIDC * HD * 2);    // [N][K]
  half_t* WoT   = (half_t*)take((size_t)HD * HIDC * 2);    // [N][K]
  half_t* Qw    = (half_t*)take((size_t)BATCHC * HC * SEQC * DC * 2);
  half_t* Kw    = (half_t*)take((size_t)BATCHC * HC * SEQC * DC * 2);
  half_t* KTs   = (half_t*)take((size_t)BATCHC * HC * SEQC * DC * 2);
  half_t* VTw   = (half_t*)take((size_t)BATCHC * HC * SEQC * DC * 2);
  half_t* Gate  = (half_t*)take((size_t)MTOT * HD * 2);
  float*  Attn  = (float*)take((size_t)MTOT * HD * 4);
  half_t* Yh    = (half_t*)take((size_t)MTOT * HD * 2);

  // 1) convert activations; transpose-convert weights once
  cvt_kernel<<<2048, 256, 0, stream>>>(hidden, Xh, (size_t)MTOT * HIDC);
  cvtT_kernel<<<dim3(NQKV / 32, HIDC / 32), 256, 0, stream>>>(w_qkv, WqkvT,
                                                              HIDC, NQKV);
  cvtT_kernel<<<dim3(HD / 32, HIDC / 32), 256, 0, stream>>>(w_gate, WgT, HIDC,
                                                            HD);
  cvtT_kernel<<<dim3(HIDC / 32, HD / 32), 256, 0, stream>>>(w_out, WoT, HD,
                                                            HIDC);

  // 2) QKV projection + silu -> Q,K row-major; decay-scaled K^T; V^T
  gemm_kernel<0><<<dim3(NQKV / 128, MTOT / 128), 256, 0, stream>>>(
      Xh, WqkvT, NQKV, Qw, Kw, KTs, VTw, nullptr);
  // 3) gate projection + sigmoid
  gemm_kernel<1><<<dim3(HD / 128, MTOT / 128), 256, 0, stream>>>(
      Xh, WgT, HD, Gate, nullptr, nullptr, nullptr, nullptr);

  // 4) lightning attention scan, one WG (16 waves) per (b, h)
  size_t smem =
      (size_t)(256 * SSTR + 64 * SSTR + 64 * SSTR + 64 * KVSTR) * 2 +
      (size_t)64 * 64 * 4;  // ~223 KB of the 320 KB WGP LDS
  (void)hipFuncSetAttribute((const void*)attn_kernel,
                            hipFuncAttributeMaxDynamicSharedMemorySize,
                            (int)smem);
  attn_kernel<<<BATCHC * HC, ATHREADS, smem, stream>>>(Qw, Kw, KTs, VTw, Attn);

  // 5) RMS norm * weight * gate
  norm_gate_kernel<<<MTOT, 256, 0, stream>>>(Attn, Gate, norm_w, Yh);

  // 6) output projection (f32 result)
  gemm_kernel<2><<<dim3(HIDC / 128, MTOT / 128), 256, 0, stream>>>(
      Yh, WoT, HIDC, nullptr, nullptr, nullptr, nullptr, out);
}
